// CrossViewAttention_18743237280507
// MI455X (gfx1250) — matrix operand order repaired and checked by reading, not compiled, not run on hardware
//
#include <hip/hip_runtime.h>
#include <hip/hip_bf16.h>

typedef __attribute__((ext_vector_type(16))) _Float16 v16h;
typedef __attribute__((ext_vector_type(8)))  _Float16 v8h;
typedef __attribute__((ext_vector_type(8)))  float    v8f;
typedef __attribute__((ext_vector_type(4)))  int      v4i;

union AFrag { v16h v; v8h h8[2]; _Float16 h[16]; };

__device__ __forceinline__ v8f wmma16(v16h a, v16h b, v8f c) {
  return __builtin_amdgcn_wmma_f32_16x16x32_f16(false, a, false, b, (short)0, c, false, false);
}

#if __has_builtin(__builtin_amdgcn_global_load_async_to_lds_b128)
#define HAVE_ASYNC_LDS 1
#else
#define HAVE_ASYNC_LDS 0
#endif

// Stage 16 bytes global -> LDS. Async path keeps the data out of the register
// file entirely (GLOBAL_LOAD_ASYNC_TO_LDS_B128, tracked by ASYNCcnt).
__device__ __forceinline__ void stage16(const _Float16* g, _Float16* l) {
#if HAVE_ASYNC_LDS
  __builtin_amdgcn_global_load_async_to_lds_b128(
      (v4i*)g,
      (__attribute__((address_space(3))) v4i*)l,
      0, 0);
#else
  *(v8h*)l = *(const v8h*)g;
#endif
}

__device__ __forceinline__ void stage_wait() {
#if HAVE_ASYNC_LDS
#if __has_builtin(__builtin_amdgcn_s_wait_asynccnt)
  __builtin_amdgcn_s_wait_asynccnt(0);
#else
  asm volatile("s_wait_asynccnt 0x0" ::: "memory");
#endif
#endif
}

// ---------------- f32 -> f16 convert ----------------
__global__ void cvt_f16_kernel(const float* __restrict__ in, _Float16* __restrict__ out, int n) {
  int i = blockIdx.x * blockDim.x + threadIdx.x;
  if (i < n) out[i] = (_Float16)in[i];
}

// ---------------- GEMM: Y = X @ W^T + b (X:[M,K] f16, W:[N,K] f16) ----------------
// X split at K0 between X0/X1 (for concat); block tile 64x64, 8 waves (4 M x 2 N),
// each wave: 16 rows x 32 cols (2 WMMA col-fragments), k-step 32 staged into LDS
// via async global->LDS loads.
template<bool RELU, bool OUT16>
__global__ __launch_bounds__(256) void gemm_wmma_kernel(
    const _Float16* __restrict__ X0, const _Float16* __restrict__ X1, int K0,
    const _Float16* __restrict__ W, const float* __restrict__ bias,
    _Float16* __restrict__ Y16, float* __restrict__ Y32,
    int M, int N, int K)
{
  __shared__ _Float16 sX[64 * 32];
  __shared__ _Float16 sW[64 * 32];
  const int t    = threadIdx.x;
  const int lane = t & 31;
  const int wav  = t >> 5;
  const int mw   = wav & 3;     // 0..3 along M
  const int nw   = wav >> 2;    // 0..1 along N
  const int m0   = blockIdx.x * 64;
  const int n0   = blockIdx.y * 64;

  const int lr = t >> 2;          // 0..63 tile row for staging
  const int lc = (t & 3) * 8;     // 0/8/16/24 halves

  const int am = lane & 15;
  const int ab = (lane < 16) ? 0 : 16;
  const int bn = lane & 15;
  const int bb = (lane < 16) ? 0 : 16;

  v8f acc0 = {}; v8f acc1 = {};

  for (int k0 = 0; k0 < K; k0 += 32) {
    __syncthreads();
    {
      int gk = k0 + lc;
      const _Float16* xs = (gk < K0) ? (X0 + (size_t)(m0 + lr) * K0 + gk)
                                     : (X1 + (size_t)(m0 + lr) * K0 + (gk - K0));
      stage16(xs, &sX[lr * 32 + lc]);
      stage16(W + (size_t)(n0 + lr) * K + gk, &sW[lr * 32 + lc]);
    }
    stage_wait();
    __syncthreads();
    AFrag a;
    a.h8[0] = *(const v8h*)&sX[(mw * 16 + am) * 32 + ab];
    a.h8[1] = *(const v8h*)&sX[(mw * 16 + am) * 32 + ab + 8];
    v16h b0 = *(const v16h*)&sW[(nw * 32 + bn) * 32 + bb];
    v16h b1 = *(const v16h*)&sW[(nw * 32 + 16 + bn) * 32 + bb];
    acc0 = wmma16(a.v, b0, acc0);
    acc1 = wmma16(a.v, b1, acc1);
  }

  const int row = (lane >> 4) * 8;
  const int col = lane & 15;
#pragma unroll
  for (int r = 0; r < 8; ++r) {
    int gm  = m0 + mw * 16 + row + r;
    int gn0 = n0 + nw * 32 + col;
    int gn1 = gn0 + 16;
    float v0 = acc0[r] + bias[gn0];
    float v1 = acc1[r] + bias[gn1];
    if (RELU) { v0 = fmaxf(v0, 0.f); v1 = fmaxf(v1, 0.f); }
    if (OUT16) {
      Y16[(size_t)gm * N + gn0] = (_Float16)v0;
      Y16[(size_t)gm * N + gn1] = (_Float16)v1;
    } else {
      Y32[(size_t)gm * N + gn0] = v0;
      Y32[(size_t)gm * N + gn1] = v1;
    }
  }
}

// ---------------- V transpose: [B][4096][512] -> [B][512][4096] (f16) ----------------
__global__ __launch_bounds__(256) void transpose_v_kernel(
    const _Float16* __restrict__ in, _Float16* __restrict__ out)
{
  __shared__ _Float16 tile[32][33];
  const int b  = blockIdx.z;
  const int n0 = blockIdx.x * 32;
  const int c0 = blockIdx.y * 32;
  const int tx = threadIdx.x & 31;
  const int ty = threadIdx.x >> 5;   // 0..7
  const _Float16* ip = in  + (size_t)b * 4096 * 512;
  _Float16*       op = out + (size_t)b * 512 * 4096;
#pragma unroll
  for (int i = 0; i < 32; i += 8)
    tile[ty + i][tx] = ip[(size_t)(n0 + ty + i) * 512 + c0 + tx];
  __syncthreads();
#pragma unroll
  for (int i = 0; i < 32; i += 8)
    op[(size_t)(c0 + ty + i) * 4096 + n0 + tx] = tile[tx][ty + i];
}

// ---------------- Attention: out = softmax(Q K^T)/sqrt(D) @ V + co ----------------
// One workgroup = 16 queries; 16x4096 f32 score panel lives in LDS (256 KB of the
// 320 KB WGP LDS). 8 waves: phase1 each owns a 512-key stripe of QK^T; phase2 exact
// softmax in LDS (P=exp(s-m) stored as f16 in place, 1/(l*sqrtD) folded into epilogue);
// phase3 each wave owns 64 output columns of P@V (V pre-transposed for contiguous B-frags).
__global__ __launch_bounds__(256) void attn_wmma_kernel(
    const _Float16* __restrict__ Q,    // [B*4096, 512]
    const _Float16* __restrict__ Km,   // [B*4096, 512]
    const _Float16* __restrict__ Vt,   // [B][512][4096]
    const float*    __restrict__ co,   // [B*4096, 512]
    float*          __restrict__ out)  // [B*4096, 512]
{
  __shared__ float sS[16 * 4096];   // 256 KB score panel (aliased by f16 P)
  __shared__ float red[256];
  __shared__ float rowL[16];

  const int t    = threadIdx.x;
  const int lane = t & 31;
  const int w    = t >> 5;
  const int q0   = blockIdx.x * 16;
  const int b    = blockIdx.y;

  const _Float16* Qb  = Q  + (size_t)b * 4096 * 512;
  const _Float16* Kb  = Km + (size_t)b * 4096 * 512;
  const _Float16* Vtb = Vt + (size_t)b * 512 * 4096;

  const int am = lane & 15;
  const int ab = (lane < 16) ? 0 : 16;
  const int bn = lane & 15;
  const int bb = (lane < 16) ? 0 : 16;
  const int row = (lane >> 4) * 8;
  const int col = lane & 15;

  // ---- phase 1: S = Q @ K^T for this wave's 512-key stripe ----
  for (int ct = 0; ct < 32; ct += 2) {
    const int c0 = w * 512 + ct * 16;
    v8f acc0 = {}; v8f acc1 = {};
#pragma unroll 4
    for (int ks = 0; ks < 16; ++ks) {
      const int d0 = ks * 32;
      AFrag a;
      a.h8[0] = *(const v8h*)&Qb[(size_t)(q0 + am) * 512 + d0 + ab];
      a.h8[1] = *(const v8h*)&Qb[(size_t)(q0 + am) * 512 + d0 + ab + 8];
      v16h b0 = *(const v16h*)&Kb[(size_t)(c0 + bn) * 512 + d0 + bb];
      v16h b1 = *(const v16h*)&Kb[(size_t)(c0 + 16 + bn) * 512 + d0 + bb];
      acc0 = wmma16(a.v, b0, acc0);
      acc1 = wmma16(a.v, b1, acc1);
    }
#pragma unroll
    for (int r = 0; r < 8; ++r) {
      sS[(row + r) * 4096 + c0 + col]      = acc0[r];
      sS[(row + r) * 4096 + c0 + 16 + col] = acc1[r];
    }
  }
  __syncthreads();

  // ---- phase 2: row-wise softmax; write P = exp(s-m) as f16 in place ----
  _Float16* sP = (_Float16*)sS;
  for (int r = 0; r < 16; ++r) {
    float e[16];
    float mx = -1e30f;
#pragma unroll
    for (int i = 0; i < 16; ++i) { e[i] = sS[r * 4096 + t + 256 * i]; mx = fmaxf(mx, e[i]); }
    red[t] = mx; __syncthreads();
    for (int s = 128; s > 0; s >>= 1) { if (t < s) red[t] = fmaxf(red[t], red[t + s]); __syncthreads(); }
    mx = red[0]; __syncthreads();
    float sum = 0.f;
#pragma unroll
    for (int i = 0; i < 16; ++i) { e[i] = __expf(e[i] - mx); sum += e[i]; }
    red[t] = sum; __syncthreads();
    for (int s = 128; s > 0; s >>= 1) { if (t < s) red[t] += red[t + s]; __syncthreads(); }
    if (t == 0) rowL[r] = red[0];
    __syncthreads();
#pragma unroll
    for (int i = 0; i < 16; ++i) sP[r * 4096 + t + 256 * i] = (_Float16)e[i];
    __syncthreads();
  }

  // ---- phase 3: O = P @ V for this wave's 64 output columns ----
  const int colb = w * 64;
  v8f o0 = {}, o1 = {}, o2 = {}, o3 = {};
  for (int k0 = 0; k0 < 4096; k0 += 32) {
    AFrag a;
    a.h8[0] = *(const v8h*)&sP[am * 4096 + k0 + ab];
    a.h8[1] = *(const v8h*)&sP[am * 4096 + k0 + ab + 8];
    v16h bf0 = *(const v16h*)&Vtb[(size_t)(colb +  0 + bn) * 4096 + k0 + bb];
    v16h bf1 = *(const v16h*)&Vtb[(size_t)(colb + 16 + bn) * 4096 + k0 + bb];
    v16h bf2 = *(const v16h*)&Vtb[(size_t)(colb + 32 + bn) * 4096 + k0 + bb];
    v16h bf3 = *(const v16h*)&Vtb[(size_t)(colb + 48 + bn) * 4096 + k0 + bb];
    o0 = wmma16(a.v, bf0, o0);
    o1 = wmma16(a.v, bf1, o1);
    o2 = wmma16(a.v, bf2, o2);
    o3 = wmma16(a.v, bf3, o3);
  }

  const float inv_sqrtD = 0.044194173824159216f; // 1/sqrt(512)
#pragma unroll
  for (int r = 0; r < 8; ++r) {
    const int m = row + r;
    const float sc = inv_sqrtD / rowL[m];
    const size_t base = ((size_t)b * 4096 + q0 + m) * 512 + colb + col;
    out[base +  0] = o0[r] * sc + co[base +  0];
    out[base + 16] = o1[r] * sc + co[base + 16];
    out[base + 32] = o2[r] * sc + co[base + 32];
    out[base + 48] = o3[r] * sc + co[base + 48];
  }
}

// ---------------- host launch ----------------
extern "C" void kernel_launch(void* const* d_in, const int* in_sizes, int n_in,
                              void* d_out, int out_size, void* d_ws, size_t ws_size,
                              hipStream_t stream) {
  const float* x_f = (const float*)d_in[0];
  const float* x_s = (const float*)d_in[1];
  const float* Wq  = (const float*)d_in[2];
  const float* bq  = (const float*)d_in[3];
  const float* Wk  = (const float*)d_in[4];
  const float* bk  = (const float*)d_in[5];
  const float* Wv  = (const float*)d_in[6];
  const float* bv  = (const float*)d_in[7];
  const float* Wc  = (const float*)d_in[8];
  const float* bc  = (const float*)d_in[9];
  float* out = (float*)d_out;

  const int Bm = 2, N = 4096, D = 512;
  const size_t M  = (size_t)Bm * N;          // 8192
  const size_t XE = M * D;                   // 4,194,304 elems

  char* ws = (char*)d_ws;
  size_t off = 0;
  auto alloc16 = [&](size_t elems) { _Float16* p = (_Float16*)(ws + off); off += elems * 2; off = (off + 255) & ~(size_t)255; return p; };
  auto alloc32 = [&](size_t elems) { float*    p = (float*)(ws + off);    off += elems * 4; off = (off + 255) & ~(size_t)255; return p; };

  _Float16* xhf = alloc16(XE);
  _Float16* xhs = alloc16(XE);
  _Float16* Wqh = alloc16((size_t)D * D);
  _Float16* Wkh = alloc16((size_t)D * D);
  _Float16* Wvh = alloc16((size_t)D * D);
  _Float16* Wch = alloc16((size_t)D * 2 * D);
  _Float16* Qf  = alloc16(XE);
  _Float16* Kf  = alloc16(XE);
  _Float16* Vf  = alloc16(XE);
  _Float16* Qs  = alloc16(XE);
  _Float16* Ks  = alloc16(XE);
  _Float16* Vs  = alloc16(XE);
  _Float16* Vtf = alloc16(XE);
  _Float16* Vts = alloc16(XE);
  float*    coW = alloc32(XE);
  (void)ws_size; (void)in_sizes; (void)n_in; (void)out_size;

  // converts
  {
    int n1 = (int)XE, n2 = D * D, n3 = D * 2 * D;
    cvt_f16_kernel<<<(n1 + 255) / 256, 256, 0, stream>>>(x_f, xhf, n1);
    cvt_f16_kernel<<<(n1 + 255) / 256, 256, 0, stream>>>(x_s, xhs, n1);
    cvt_f16_kernel<<<(n2 + 255) / 256, 256, 0, stream>>>(Wq, Wqh, n2);
    cvt_f16_kernel<<<(n2 + 255) / 256, 256, 0, stream>>>(Wk, Wkh, n2);
    cvt_f16_kernel<<<(n2 + 255) / 256, 256, 0, stream>>>(Wv, Wvh, n2);
    cvt_f16_kernel<<<(n3 + 255) / 256, 256, 0, stream>>>(Wc, Wch, n3);
  }

  // projections: Y = X @ W^T + b  (f16 out)
  dim3 ggrid((unsigned)(M / 64), (unsigned)(D / 64));
  gemm_wmma_kernel<false, true><<<ggrid, 256, 0, stream>>>(xhf, xhf, D, Wqh, bq, Qf, nullptr, (int)M, D, D);
  gemm_wmma_kernel<false, true><<<ggrid, 256, 0, stream>>>(xhf, xhf, D, Wkh, bk, Kf, nullptr, (int)M, D, D);
  gemm_wmma_kernel<false, true><<<ggrid, 256, 0, stream>>>(xhf, xhf, D, Wvh, bv, Vf, nullptr, (int)M, D, D);
  gemm_wmma_kernel<false, true><<<ggrid, 256, 0, stream>>>(xhs, xhs, D, Wqh, bq, Qs, nullptr, (int)M, D, D);
  gemm_wmma_kernel<false, true><<<ggrid, 256, 0, stream>>>(xhs, xhs, D, Wkh, bk, Ks, nullptr, (int)M, D, D);
  gemm_wmma_kernel<false, true><<<ggrid, 256, 0, stream>>>(xhs, xhs, D, Wvh, bv, Vs, nullptr, (int)M, D, D);
  // co_occur = relu([x_f|x_s] @ Wc^T + bc)  (f32 out)
  gemm_wmma_kernel<true, false><<<ggrid, 256, 0, stream>>>(xhf, xhs, D, Wch, bc, nullptr, coW, (int)M, D, 2 * D);

  // V transposes for contiguous PV B-fragments
  dim3 tgrid(N / 32, D / 32, Bm);
  transpose_v_kernel<<<tgrid, 256, 0, stream>>>(Vf, Vtf);
  transpose_v_kernel<<<tgrid, 256, 0, stream>>>(Vs, Vts);

  // attention (one workgroup per 16 queries; 256 KB LDS score panel)
  dim3 agrid(N / 16, Bm);
  attn_wmma_kernel<<<agrid, 256, 0, stream>>>(Qf, Ks, Vtf, coW, out);                 // out_f
  attn_wmma_kernel<<<agrid, 256, 0, stream>>>(Qs, Kf, Vts, coW, out + (size_t)Bm * N * D); // out_s
}
